// TaskAlignedAssigner_54485955117105
// MI455X (gfx1250) — matrix operand order repaired and checked by reading, not compiled, not run on hardware
//
#include <hip/hip_runtime.h>
#include <hip/hip_bf16.h>

#define BS  32
#define AN  8400
#define NC  80
#define MG  64
#define TK  10
#define ATILES 5          // anchor tiles (of 16) per wave in K1; 525 = 105 * 5

typedef __attribute__((ext_vector_type(16))) _Float16 v16h;
typedef __attribute__((ext_vector_type(8)))  float    v8f;

// d_out layout (floats, reference tuple order)
static constexpr size_t O_LABELS = 0;
static constexpr size_t O_BBOXES = (size_t)BS * AN;
static constexpr size_t O_SCORES = O_BBOXES + (size_t)BS * AN * 4;
static constexpr size_t O_FG     = O_SCORES + (size_t)BS * AN * NC;
static constexpr size_t O_TGT    = O_FG + (size_t)BS * AN;

// workspace layout (bytes)
static constexpr size_t AMET_BYTES = (size_t)BS * MG * AN * sizeof(float);         // 68.8 MB
static constexpr size_t MP_BYTES   = (size_t)BS * AN * sizeof(unsigned long long); // 2.15 MB

__device__ __forceinline__ float ciou_clipped(float gx1, float gy1, float gx2, float gy2,
                                              float px1, float py1, float px2, float py2)
{
    const float eps = 1e-7f;
    const float w1 = gx2 - gx1, h1 = gy2 - gy1 + eps;   // b1 = gt
    const float w2 = px2 - px1, h2 = py2 - py1 + eps;   // b2 = pred
    float iw = fminf(gx2, px2) - fmaxf(gx1, px1); iw = fmaxf(iw, 0.f);
    float ih = fminf(gy2, py2) - fmaxf(gy1, py1); ih = fmaxf(ih, 0.f);
    const float inter = iw * ih;
    const float uni   = w1 * h1 + w2 * h2 - inter + eps;
    const float iou   = inter / uni;
    const float cw = fmaxf(gx2, px2) - fminf(gx1, px1);
    const float ch = fmaxf(gy2, py2) - fminf(gy1, py1);
    const float c2 = cw * cw + ch * ch + eps;
    const float dx = px1 + px2 - gx1 - gx2;
    const float dy = py1 + py2 - gy1 - gy2;
    const float rho2 = (dx * dx + dy * dy) * 0.25f;
    const float pi = 3.14159265358979323846f;
    const float dv = atanf(w2 / h2) - atanf(w1 / h1);
    const float v  = (4.0f / (pi * pi)) * dv * dv;
    const float alpha = v / (v - iou + (1.0f + eps));
    return fmaxf(iou - (rho2 / c2 + v * alpha), 0.0f);   // clip(ciou, 0)
}

// K1: fused WMMA one-hot gather (bbox_scores) + CIoU -> align_metric (unmasked).
// One wave per block; tile = 16 gts x 16 anchors; K loop over C=80 in 3 chunks of 32.
// A tiles (one-hot of labels) and gt boxes are hoisted across ATILES anchor tiles.
__global__ __launch_bounds__(32)
void k_align_metric(const float* __restrict__ scores,
                    const float* __restrict__ pdb,
                    const float* __restrict__ gtb,
                    const int*   __restrict__ glab,
                    float* __restrict__ amet)
{
    const int lane  = threadIdx.x;
    const int col   = lane & 15;
    const int hi    = lane >> 4;                 // half-wave select
    const int abase = blockIdx.x * (ATILES * 16);
    const int m0    = blockIdx.y * 16;
    const int b     = blockIdx.z;

    const int lab = glab[b * MG + m0 + col];     // A-matrix row m = m0+col

    // Pre-build one-hot A tiles: 16x32 f16 tiles, this code's k order k = e + 16*hi (same as B)
    v16h av[3];
#pragma unroll
    for (int cb = 0; cb < 3; ++cb) {
        const int cbase = cb * 32;
#pragma unroll
        for (int e = 0; e < 16; ++e) {
            const int ka = (hi << 4) + e;
            av[cb][e] = (_Float16)((cbase + ka == lab) ? 1.f : 0.f);
        }
    }
    // gt boxes for this lane's 8 output rows: m = m0 + r + 8*hi
    float4 gbs[8];
#pragma unroll
    for (int r = 0; r < 8; ++r)
        gbs[r] = reinterpret_cast<const float4*>(gtb)[b * MG + m0 + r + (hi << 3)];

    for (int t = 0; t < ATILES; ++t) {
        const int    a    = abase + t * 16 + col;
        const float* srow = scores + ((size_t)b * AN + a) * NC;

        v8f acc = {};
#pragma unroll
        for (int cb = 0; cb < 3; ++cb) {
            // B (16-bit 32x16): lane holds column col; this code's k order k = e + 16*hi, as in A
            v16h bv;
            const int cstart = cb * 32 + (hi << 4);
            if (cstart < NC) {
                const float4* p = reinterpret_cast<const float4*>(srow + cstart);
#pragma unroll
                for (int q = 0; q < 4; ++q) {
                    float4 f = p[q];
                    bv[4*q + 0] = (_Float16)f.x; bv[4*q + 1] = (_Float16)f.y;
                    bv[4*q + 2] = (_Float16)f.z; bv[4*q + 3] = (_Float16)f.w;
                }
            } else {
#pragma unroll
                for (int e = 0; e < 16; ++e) bv[e] = (_Float16)0.f;
            }
            acc = __builtin_amdgcn_wmma_f32_16x16x32_f16(false, av[cb], false, bv,
                                                         (short)0, acc, false, false);
        }

        // D layout: acc[r] = bbox_score for (m = m0 + r + 8*hi, anchor a)
        const float4 pb = reinterpret_cast<const float4*>(pdb)[(size_t)b * AN + a];
#pragma unroll
        for (int r = 0; r < 8; ++r) {
            const int   m  = m0 + r + (hi << 3);
            const float ov = ciou_clipped(gbs[r].x, gbs[r].y, gbs[r].z, gbs[r].w,
                                          pb.x, pb.y, pb.z, pb.w);
            amet[((size_t)(b * MG + m)) * AN + a] = acc[r] * ov;
        }
    }
}

// K2: per (b,m) top-10 of align_metric * in_gts over A anchors -> set bits in mask_pos.
// Single streaming pass with register-resident per-thread top-10, then merge in LDS.
__global__ __launch_bounds__(256)
void k_topk(const float* __restrict__ amet, const float* __restrict__ anc,
            const float* __restrict__ gtb,  const float* __restrict__ maskgt,
            unsigned long long* __restrict__ mp)
{
    const int m = blockIdx.x, b = blockIdx.y, tid = threadIdx.x;
    if (maskgt[b * MG + m] <= 0.f) return;   // invalid gt: contributes no bits

    __shared__ float sv[256 * TK];
    __shared__ int   si[256 * TK];
    __shared__ float rv[256];
    __shared__ int   ri[256];
    __shared__ int   rp[256];

    const float4 gb   = reinterpret_cast<const float4*>(gtb)[b * MG + m];
    const float* arow = amet + ((size_t)(b * MG + m)) * AN;

    // per-thread sorted (desc) top-10; ascending scan + strict '>' => lower index wins ties
    float tv[TK]; int ti[TK];
#pragma unroll
    for (int i = 0; i < TK; ++i) { tv[i] = -1e30f; ti[i] = 0x7fffffff; }

    for (int a = tid; a < AN; a += 256) {
        __builtin_prefetch(arow + a + 2048, 0, 0);
        const float gx = anc[2 * a], gy = anc[2 * a + 1];
        const float d  = fminf(fminf(gx - gb.x, gy - gb.y), fminf(gb.z - gx, gb.w - gy));
        const float v  = (d > 1e-9f) ? arow[a] : 0.f;   // selection metric (masked)
        if (v > tv[TK - 1]) {
            tv[TK - 1] = v; ti[TK - 1] = a;
#pragma unroll
            for (int j = TK - 1; j > 0; --j) {
                if (tv[j] > tv[j - 1]) {
                    const float fv = tv[j]; tv[j] = tv[j - 1]; tv[j - 1] = fv;
                    const int   iv = ti[j]; ti[j] = ti[j - 1]; ti[j - 1] = iv;
                }
            }
        }
    }
#pragma unroll
    for (int i = 0; i < TK; ++i) { sv[tid * TK + i] = tv[i]; si[tid * TK + i] = ti[i]; }
    __syncthreads();

    // 10 argmax rounds over the 2560 merged candidates (value desc, index asc)
    for (int it = 0; it < TK; ++it) {
        float bvv = -1e30f; int bii = 0x7fffffff; int bpp = 0;
        for (int e = tid; e < 256 * TK; e += 256) {
            const float v = sv[e]; const int ix = si[e];
            if (v > bvv || (v == bvv && ix < bii)) { bvv = v; bii = ix; bpp = e; }
        }
        rv[tid] = bvv; ri[tid] = bii; rp[tid] = bpp;
        __syncthreads();
        for (int s = 128; s > 0; s >>= 1) {
            if (tid < s) {
                const float ov = rv[tid + s]; const int oi = ri[tid + s];
                if (ov > rv[tid] || (ov == rv[tid] && oi < ri[tid])) {
                    rv[tid] = ov; ri[tid] = oi; rp[tid] = rp[tid + s];
                }
            }
            __syncthreads();
        }
        if (tid == 0) {
            sv[rp[0]] = -2e30f;                          // consume candidate
            const int idx = ri[0];
            const float gx = anc[2 * idx], gy = anc[2 * idx + 1];
            const float d  = fminf(fminf(gx - gb.x, gy - gb.y), fminf(gb.z - gx, gb.w - gy));
            if (d > 1e-9f)                               // mask_topk * in_gts * mask_gt
                atomicOr(&mp[(size_t)b * AN + idx], 1ull << m);
        }
        __syncthreads();
    }
}

// K3: per (b,a) conflict resolution + labels/bboxes/fg/tgt outputs
__global__ __launch_bounds__(256)
void k_resolve(const float* __restrict__ pdb, const float* __restrict__ gtb,
               const int* __restrict__ glab, unsigned long long* __restrict__ mp,
               float* __restrict__ out)
{
    const int b = blockIdx.y;
    const int a = blockIdx.x * 256 + threadIdx.x;
    __shared__ float4 sgb[MG];
    __shared__ int    slab[MG];
    if (threadIdx.x < MG) {
        sgb[threadIdx.x]  = reinterpret_cast<const float4*>(gtb)[b * MG + threadIdx.x];
        slab[threadIdx.x] = glab[b * MG + threadIdx.x];
    }
    __syncthreads();
    if (a >= AN) return;

    const size_t ba = (size_t)b * AN + a;
    unsigned long long bits = mp[ba];
    const int fgc = __popcll(bits);                     // fg_count (pre-substitution)
    int tgt = 0; float fg = 0.f;
    if (fgc == 1) {
        tgt = __ffsll((long long)bits) - 1; fg = 1.f;
    } else if (fgc > 1) {
        // replace column with one-hot of argmax_m overlaps (all m, first-max tiebreak)
        const float4 pb = reinterpret_cast<const float4*>(pdb)[ba];
        float best = -1.f; int bi = 0;
        for (int m = 0; m < MG; ++m) {
            const float4 g = sgb[m];
            const float  o = ciou_clipped(g.x, g.y, g.z, g.w, pb.x, pb.y, pb.z, pb.w);
            if (o > best) { best = o; bi = m; }
        }
        tgt = bi; fg = 1.f;
        mp[ba] = 1ull << bi;                            // final mask_pos
    }
    out[O_LABELS + ba] = (float)slab[tgt];
    const float4 tb = sgb[tgt];
    float* ob = out + O_BBOXES + ba * 4;
    ob[0] = tb.x; ob[1] = tb.y; ob[2] = tb.z; ob[3] = tb.w;
    out[O_FG  + ba] = fg;
    out[O_TGT + ba] = (float)tgt;
}

// K4: per (b,m) pos_align / pos_overlaps maxima over final mask_pos
__global__ __launch_bounds__(256)
void k_posmax(const float* __restrict__ amet, const float* __restrict__ pdb,
              const float* __restrict__ gtb, const unsigned long long* __restrict__ mp,
              float* __restrict__ posal, float* __restrict__ posov)
{
    const int m = blockIdx.x, b = blockIdx.y, tid = threadIdx.x;
    const float4 gb   = reinterpret_cast<const float4*>(gtb)[b * MG + m];
    const float* arow = amet + ((size_t)(b * MG + m)) * AN;
    const unsigned long long bit = 1ull << m;
    float ma = 0.f, mo = 0.f;
    for (int a = tid; a < AN; a += 256) {
        if (mp[(size_t)b * AN + a] & bit) {
            ma = fmaxf(ma, arow[a]);
            const float4 pb = reinterpret_cast<const float4*>(pdb)[(size_t)b * AN + a];
            mo = fmaxf(mo, ciou_clipped(gb.x, gb.y, gb.z, gb.w, pb.x, pb.y, pb.z, pb.w));
        }
    }
    __shared__ float s1[256], s2[256];
    s1[tid] = ma; s2[tid] = mo;
    __syncthreads();
    for (int s = 128; s > 0; s >>= 1) {
        if (tid < s) { s1[tid] = fmaxf(s1[tid], s1[tid + s]); s2[tid] = fmaxf(s2[tid], s2[tid + s]); }
        __syncthreads();
    }
    if (tid == 0) { posal[b * MG + m] = s1[0]; posov[b * MG + m] = s2[0]; }
}

// K5: per (b,a) norm = max_m amet*bit*pos_ovl/(pos_align+EPS); scatter into zeroed scores
__global__ __launch_bounds__(256)
void k_norm_scatter(const float* __restrict__ amet, const unsigned long long* __restrict__ mp,
                    const float* __restrict__ posal, const float* __restrict__ posov,
                    float* __restrict__ out)
{
    const int b = blockIdx.y;
    const int a = blockIdx.x * 256 + threadIdx.x;
    __shared__ float sal[MG], sov[MG];
    if (threadIdx.x < MG) {
        sal[threadIdx.x] = posal[b * MG + threadIdx.x];
        sov[threadIdx.x] = posov[b * MG + threadIdx.x];
    }
    __syncthreads();
    if (a >= AN) return;

    const size_t ba = (size_t)b * AN + a;
    unsigned long long bits = mp[ba];
    float norm = 0.f;
    while (bits) {
        const int m = __ffsll((long long)bits) - 1;
        bits &= bits - 1;
        norm = fmaxf(norm, amet[((size_t)(b * MG + m)) * AN + a] * sov[m] / (sal[m] + 1e-9f));
    }
    if (out[O_FG + ba] > 0.f) {
        const int lab = (int)out[O_LABELS + ba];
        out[O_SCORES + ba * NC + lab] = norm;
    }
}

extern "C" void kernel_launch(void* const* d_in, const int* in_sizes, int n_in,
                              void* d_out, int out_size, void* d_ws, size_t ws_size,
                              hipStream_t stream)
{
    (void)in_sizes; (void)n_in; (void)out_size; (void)ws_size;
    const float* pd_scores = (const float*)d_in[0];
    const float* pd_bboxes = (const float*)d_in[1];
    const float* anc       = (const float*)d_in[2];
    const int*   gt_labels = (const int*)d_in[3];
    const float* gt_bboxes = (const float*)d_in[4];
    const float* mask_gt   = (const float*)d_in[5];
    float* out = (float*)d_out;

    char* ws = (char*)d_ws;
    float*              amet  = (float*)ws;
    unsigned long long* mp    = (unsigned long long*)(ws + AMET_BYTES);
    float*              posal = (float*)(ws + AMET_BYTES + MP_BYTES);
    float*              posov = posal + BS * MG;

    hipMemsetAsync(mp, 0, MP_BYTES, stream);
    hipMemsetAsync(out + O_SCORES, 0, (size_t)BS * AN * NC * sizeof(float), stream);

    k_align_metric<<<dim3(AN / (16 * ATILES), MG / 16, BS), 32, 0, stream>>>(
        pd_scores, pd_bboxes, gt_bboxes, gt_labels, amet);
    k_topk<<<dim3(MG, BS), 256, 0, stream>>>(amet, anc, gt_bboxes, mask_gt, mp);
    k_resolve<<<dim3((AN + 255) / 256, BS), 256, 0, stream>>>(
        pd_bboxes, gt_bboxes, gt_labels, mp, out);
    k_posmax<<<dim3(MG, BS), 256, 0, stream>>>(amet, pd_bboxes, gt_bboxes, mp, posal, posov);
    k_norm_scatter<<<dim3((AN + 255) / 256, BS), 256, 0, stream>>>(amet, mp, posal, posov, out);
}
